// NNUEModelSparse_50560355008950
// MI455X (gfx1250) — matrix operand order repaired and checked by reading, not compiled, not run on hardware
//
#include <hip/hip_runtime.h>
#include <hip/hip_bf16.h>

typedef __attribute__((ext_vector_type(16))) _Float16 v16h;
typedef __attribute__((ext_vector_type(8)))  _Float16 v8h;
typedef __attribute__((ext_vector_type(4)))  _Float16 v4h;
typedef __attribute__((ext_vector_type(8)))  float    v8f;

#define BATCH    16384
#define NNZ_PER  32
#define NNZ      (BATCH * NNZ_PER)
#define HIDDEN   256
#define IN2      512   // 2*HIDDEN
#define H2D      32
#define H3D      32

// ---------------------------------------------------------------------------
// Phase 1: sparse feature transform (L2-gather bound; ft_w is 42MB and lives
// in the 192MB L2).
// 64 threads own one 256-wide accumulator row as float4 lanes; each feature
// gather is one global_load_b128 per thread (coalesced 1KB row burst).
// Block = 512 threads = 8 row-groups = 4 boards (white+black per board).
// Output x[b][0:256]=crelu(white), x[b][256:512]=crelu(black), packed f16.
// ---------------------------------------------------------------------------
__global__ __launch_bounds__(512) void nnue_ft(
    const int* __restrict__ wfeat, const float* __restrict__ wval,
    const int* __restrict__ bfeat, const float* __restrict__ bval,
    const float* __restrict__ ftw, const float* __restrict__ ftb,
    _Float16* __restrict__ x)
{
  const int tid  = threadIdx.x;
  const int q    = tid & 63;             // float4 slot within the 256-row
  const int rg   = tid >> 6;             // 0..7: row-group within block
  const int side = rg & 1;               // 0 = white, 1 = black
  const int b    = blockIdx.x * 4 + (rg >> 1);

  const int*   feat = side ? bfeat : wfeat;   // already offset to feat row
  const float* val  = side ? bval  : wval;

  const float4* ftw4 = (const float4*)ftw;    // [40960][64] float4
  const float4* ftb4 = (const float4*)ftb;

  float4 acc = ftb4[q];
  const int base = b * NNZ_PER;
  #pragma unroll 8
  for (int k = 0; k < NNZ_PER; ++k) {
    const int    f = feat[base + k];          // wave-uniform index
    const float  v = val[base + k];
    const float4 w = ftw4[(size_t)f * (HIDDEN / 4) + q];
    acc.x = fmaf(w.x, v, acc.x);
    acc.y = fmaf(w.y, v, acc.y);
    acc.z = fmaf(w.z, v, acc.z);
    acc.w = fmaf(w.w, v, acc.w);
  }
  v4h o;                                       // crelu -> [0,1], f16-safe
  o[0] = (_Float16)fminf(fmaxf(acc.x, 0.0f), 1.0f);
  o[1] = (_Float16)fminf(fmaxf(acc.y, 0.0f), 1.0f);
  o[2] = (_Float16)fminf(fmaxf(acc.z, 0.0f), 1.0f);
  o[3] = (_Float16)fminf(fmaxf(acc.w, 0.0f), 1.0f);
  *(v4h*)&x[(size_t)b * IN2 + side * HIDDEN + q * 4] = o;  // 8B store
}

// ---------------------------------------------------------------------------
// Phase 2: dense head with WMMA f16.
//   fc1: [16x512] x [512x32]  -> 16x f16 WMMA(K=32) per N-tile (2 tiles)
//   fc2: [16x32]  x [32x32]   -> 1x WMMA per N-tile (via per-wave LDS stage)
//   fc3: [16x32]  x [32x1]    -> shfl_xor reduction over 16-lane halves
// Block = 256 threads (8 waves), one 16-row batch tile per wave.
// Weights pre-converted to f16 in LDS, transposed to [n][K] so every B
// fragment is two contiguous ds_load_b128.
// ---------------------------------------------------------------------------
__global__ __launch_bounds__(256) void nnue_mlp(
    const _Float16* __restrict__ x,
    const float* __restrict__ fc1w, const float* __restrict__ fc1b,
    const float* __restrict__ fc2w, const float* __restrict__ fc2b,
    const float* __restrict__ fc3w, const float* __restrict__ fc3b,
    float* __restrict__ out)
{
  __shared__ _Float16 sW1[H2D * IN2];     // fc1_w transposed: [n][K], f16
  __shared__ _Float16 sW2[H3D * H2D];     // fc2_w transposed: [n][K], f16
  __shared__ _Float16 sH1[8][16 * H2D];   // per-wave crelu(fc1) staging
  __shared__ float    sB1[H2D], sB2[H3D], sW3[H3D];

  const int tid = threadIdx.x;
  for (int i = tid; i < H2D * IN2; i += 256) {
    const int K = i >> 5, n = i & 31;
    sW1[n * IN2 + K] = (_Float16)fc1w[i];
  }
  for (int i = tid; i < H3D * H2D; i += 256) {
    const int K = i >> 5, n = i & 31;
    sW2[n * H2D + K] = (_Float16)fc2w[i];
  }
  if (tid < H2D) { sB1[tid] = fc1b[tid]; sB2[tid] = fc2b[tid]; sW3[tid] = fc3w[tid]; }
  __syncthreads();

  const int wave   = tid >> 5;
  const int lane   = tid & 31;
  const int lan16  = lane & 15;
  const int hiHalf = lane >> 4;                  // 0: lanes 0-15, 1: lanes 16-31
  const int row0   = (blockIdx.x * 8 + wave) * 16;
  const int kcA    = hiHalf * 8;                 // A frag: K = kcA+{0..7,16..23}
  const int kcB    = hiHalf * 8;                 // B frag: K = kcB+{0..7,16..23}
  const int mbase  = hiHalf * 8;                 // C frag: M = mbase + r

  // ---- fc1 ----
  v8f c0, c1;
  #pragma unroll
  for (int r = 0; r < 8; ++r) { c0[r] = sB1[lan16]; c1[r] = sB1[16 + lan16]; }

  const _Float16* xrow = x + (size_t)(row0 + lan16) * IN2;
  #pragma unroll 4
  for (int kb = 0; kb < IN2 / 32; ++kb) {
    const int k0 = kb * 32;
    v8h alo = *(const v8h*)(xrow + k0 + kcA);
    v8h ahi = *(const v8h*)(xrow + k0 + kcA + 16);
    v16h a;
    #pragma unroll
    for (int i = 0; i < 8; ++i) { a[i] = alo[i]; a[8 + i] = ahi[i]; }

    const _Float16* b0p = &sW1[lan16 * IN2 + k0 + kcB];
    const _Float16* b1p = &sW1[(16 + lan16) * IN2 + k0 + kcB];
    v8h b0lo = *(const v8h*)b0p, b0hi = *(const v8h*)(b0p + 16);
    v8h b1lo = *(const v8h*)b1p, b1hi = *(const v8h*)(b1p + 16);
    v16h bf0, bf1;
    #pragma unroll
    for (int i = 0; i < 8; ++i) {
      bf0[i] = b0lo[i]; bf0[8 + i] = b0hi[i];
      bf1[i] = b1lo[i]; bf1[8 + i] = b1hi[i];
    }

    c0 = __builtin_amdgcn_wmma_f32_16x16x32_f16(false, a, false, bf0, (short)0, c0, false, false);
    c1 = __builtin_amdgcn_wmma_f32_16x16x32_f16(false, a, false, bf1, (short)0, c1, false, false);
  }

  // crelu + stage h1 per-wave (same-wave DS ops are in-order: no barrier)
  _Float16* h1 = &sH1[wave][0];
  #pragma unroll
  for (int r = 0; r < 8; ++r) {
    const float v0 = fminf(fmaxf(c0[r], 0.0f), 1.0f);
    const float v1 = fminf(fmaxf(c1[r], 0.0f), 1.0f);
    h1[(mbase + r) * H2D + lan16]      = (_Float16)v0;
    h1[(mbase + r) * H2D + 16 + lan16] = (_Float16)v1;
  }

  // ---- fc2 ----
  v16h a2;
  {
    const _Float16* ap = &h1[lan16 * H2D];
    v8h alo = *(const v8h*)(ap + kcA);
    v8h ahi = *(const v8h*)(ap + kcA + 16);
    #pragma unroll
    for (int i = 0; i < 8; ++i) { a2[i] = alo[i]; a2[8 + i] = ahi[i]; }
  }
  v16h b20, b21;
  {
    const _Float16* b0p = &sW2[lan16 * H2D + kcB];
    const _Float16* b1p = &sW2[(16 + lan16) * H2D + kcB];
    v8h b0lo = *(const v8h*)b0p, b0hi = *(const v8h*)(b0p + 16);
    v8h b1lo = *(const v8h*)b1p, b1hi = *(const v8h*)(b1p + 16);
    #pragma unroll
    for (int i = 0; i < 8; ++i) {
      b20[i] = b0lo[i]; b20[8 + i] = b0hi[i];
      b21[i] = b1lo[i]; b21[8 + i] = b1hi[i];
    }
  }
  v8f d0, d1;
  #pragma unroll
  for (int r = 0; r < 8; ++r) { d0[r] = sB2[lan16]; d1[r] = sB2[16 + lan16]; }
  d0 = __builtin_amdgcn_wmma_f32_16x16x32_f16(false, a2, false, b20, (short)0, d0, false, false);
  d1 = __builtin_amdgcn_wmma_f32_16x16x32_f16(false, a2, false, b21, (short)0, d1, false, false);

  // ---- fc3: per-row dot over 32 cols spread across 16 lanes ----
  const float w3a = sW3[lan16], w3b = sW3[16 + lan16];
  const float b3  = fc3b[0];
  #pragma unroll
  for (int r = 0; r < 8; ++r) {
    const float h0  = fminf(fmaxf(d0[r], 0.0f), 1.0f);
    const float h1v = fminf(fmaxf(d1[r], 0.0f), 1.0f);
    float p = h0 * w3a + h1v * w3b;
    p += __shfl_xor(p, 1);
    p += __shfl_xor(p, 2);
    p += __shfl_xor(p, 4);
    p += __shfl_xor(p, 8);       // groups of 16 lanes stay separate
    if (lan16 == 0) out[row0 + mbase + r] = p + b3;
  }
}

// ---------------------------------------------------------------------------
extern "C" void kernel_launch(void* const* d_in, const int* in_sizes, int n_in,
                              void* d_out, int out_size, void* d_ws, size_t ws_size,
                              hipStream_t stream) {
  const int*   wi   = (const int*)d_in[0];    // white_indices [2, NNZ]
  const float* wv   = (const float*)d_in[1];  // white_values  [NNZ]
  const int*   bi   = (const int*)d_in[2];    // black_indices [2, NNZ]
  const float* bv   = (const float*)d_in[3];  // black_values  [NNZ]
  const float* ftw  = (const float*)d_in[4];  // [40960, 256]
  const float* ftb  = (const float*)d_in[5];  // [256]
  const float* fc1w = (const float*)d_in[6];  // [512, 32]
  const float* fc1b = (const float*)d_in[7];  // [32]
  const float* fc2w = (const float*)d_in[8];  // [32, 32]
  const float* fc2b = (const float*)d_in[9];  // [32]
  const float* fc3w = (const float*)d_in[10]; // [32, 1]
  const float* fc3b = (const float*)d_in[11]; // [1]
  float* out = (float*)d_out;                 // [16384, 1]

  _Float16* x = (_Float16*)d_ws;              // [16384, 512] f16, 16 MB

  // row 0 of indices is batch index (b*32..b*32+31 by construction);
  // row 1 (offset NNZ) is the feature index.
  nnue_ft<<<BATCH / 4, 512, 0, stream>>>(wi + NNZ, wv, bi + NNZ, bv, ftw, ftb, x);
  nnue_mlp<<<BATCH / 128, 256, 0, stream>>>(x, fc1w, fc1b, fc2w, fc2b, fc3w, fc3b, out);
}
